// CausalSelfAttention_14637248545207
// MI455X (gfx1250) — matrix operand order repaired and checked by reading, not compiled, not run on hardware
//
#include <hip/hip_runtime.h>

// ---------------------------------------------------------------------------
// CausalSelfAttention for gfx1250 (MI455X): bf16 WMMA everywhere.
// B=4, S=2048, E=1024, H=16, D=64.
// ---------------------------------------------------------------------------

typedef __attribute__((ext_vector_type(16))) __bf16 v16bf;
typedef __attribute__((ext_vector_type(8)))  __bf16 v8bf;
typedef __attribute__((ext_vector_type(4)))  __bf16 v4bf;
typedef __attribute__((ext_vector_type(8)))  float  v8f;

#define NB   4
#define NS   2048
#define NE   1024
#define NH   16
#define ND   64
#define NM   (NB * NS)     // 8192 rows
#define N3E  (3 * NE)      // 3072

// ---------------------------------------------------------------------------
// fp32 -> bf16 conversion (vectorized: 4 elems / thread)
// ---------------------------------------------------------------------------
__global__ __launch_bounds__(256) void cvt_f32_bf16_4(const float* __restrict__ in,
                                                      __bf16* __restrict__ out, int n4) {
    int i = blockIdx.x * blockDim.x + threadIdx.x;
    if (i >= n4) return;
    float4 f = ((const float4*)in)[i];
    v4bf r = { (__bf16)f.x, (__bf16)f.y, (__bf16)f.z, (__bf16)f.w };
    ((v4bf*)out)[i] = r;
}

// ---------------------------------------------------------------------------
// WMMA fragment loaders (bf16, row-major source, contraction dim contiguous)
// A-matrix 16x32 layout: lane L -> M=L&15; elems 0..15 = K[k0 + (L>>4)*16 ..]
// B-matrix 32x16 layout: lane L -> N=L&15; elems 0..15 = K[k0 + (L>>4)*16 ..]
// ---------------------------------------------------------------------------
__device__ __forceinline__ v16bf load_a_frag(const __bf16* __restrict__ base, int ld,
                                             int row0, int k0, int lane) {
    int m = lane & 15, hl = lane >> 4;
    const __bf16* p = base + (size_t)(row0 + m) * ld + k0 + hl * 16;
    v8bf lo = *(const v8bf*)p;
    v8bf hi = *(const v8bf*)(p + 8);
    return __builtin_shufflevector(lo, hi, 0,1,2,3,4,5,6,7,8,9,10,11,12,13,14,15);
}

__device__ __forceinline__ v16bf load_b_frag(const __bf16* __restrict__ base, int ld,
                                             int row0, int k0, int lane) {
    int n = lane & 15, hl = lane >> 4;
    const __bf16* p = base + (size_t)(row0 + n) * ld + k0 + hl * 16;
    v8bf lo = *(const v8bf*)p;
    v8bf hi = *(const v8bf*)(p + 8);
    return __builtin_shufflevector(lo, hi, 0,1,2,3,4,5,6,7,8,9,10,11,12,13,14,15);
}

__device__ __forceinline__ v8f wmma_bf16(v16bf a, v16bf b, v8f c) {
    return __builtin_amdgcn_wmma_f32_16x16x32_bf16(false, a, false, b, (short)0, c,
                                                   false, false);
}

// ---------------------------------------------------------------------------
// QKV GEMM: qkv[m,f] = sum_e Xb[m,e]*Wb[f,e] + bias[f]    (NT GEMM)
// 64x64 tile per wave. Writes Q,K as [B,H,S,D] bf16 and V transposed [B,H,D,S].
// ---------------------------------------------------------------------------
__global__ __launch_bounds__(256) void qkv_gemm(const __bf16* __restrict__ X,
                                                const __bf16* __restrict__ W,
                                                const float* __restrict__ bias,
                                                __bf16* __restrict__ Qb,
                                                __bf16* __restrict__ Kb,
                                                __bf16* __restrict__ VTb) {
    const int lane = threadIdx.x & 31;
    const int wid  = threadIdx.x >> 5;
    const int tid  = blockIdx.x * 8 + wid;           // 6144 wave tiles
    const int tilesN = N3E / 64;                     // 48
    const int tn = tid % tilesN, tm = tid / tilesN;
    const int m0 = tm * 64, n0 = tn * 64;

    const v8f z8 = {0.f,0.f,0.f,0.f,0.f,0.f,0.f,0.f};
    v8f acc[4][4];
    #pragma unroll
    for (int i = 0; i < 4; i++)
        #pragma unroll
        for (int j = 0; j < 4; j++) acc[i][j] = z8;

    for (int k0 = 0; k0 < NE; k0 += 32) {
        if (k0 + 32 < NE) {
            __builtin_prefetch(X + (size_t)(m0 + (lane & 15)) * NE + k0 + 32, 0, 1);
            __builtin_prefetch(W + (size_t)(n0 + (lane & 15)) * NE + k0 + 32, 0, 1);
        }
        v16bf a[4], b[4];
        #pragma unroll
        for (int i = 0; i < 4; i++) a[i] = load_a_frag(X, NE, m0 + 16 * i, k0, lane);
        #pragma unroll
        for (int j = 0; j < 4; j++) b[j] = load_b_frag(W, NE, n0 + 16 * j, k0, lane);
        #pragma unroll
        for (int i = 0; i < 4; i++)
            #pragma unroll
            for (int j = 0; j < 4; j++)
                acc[i][j] = wmma_bf16(a[i], b[j], acc[i][j]);
    }

    // Epilogue: scatter into Q / K / V^T (bf16)
    const int nl = lane & 15, hl = lane >> 4;
    #pragma unroll
    for (int j = 0; j < 4; j++) {
        const int n = n0 + 16 * j + nl;
        const float bn = bias[n];
        const int which = n >> 10;          // 0=Q 1=K 2=V (uniform per wave/tile)
        const int rem = n & 1023;
        const int hh = rem >> 6, d = rem & 63;
        #pragma unroll
        for (int i = 0; i < 4; i++) {
            #pragma unroll
            for (int r = 0; r < 8; r++) {
                const int m = m0 + 16 * i + r + 8 * hl;
                const int bb = m >> 11, s = m & 2047;
                const __bf16 bv = (__bf16)(acc[i][j][r] + bn);
                const size_t bh = (size_t)(bb * NH + hh);
                if (which == 0)      Qb[(bh * NS + s) * ND + d] = bv;
                else if (which == 1) Kb[(bh * NS + s) * ND + d] = bv;
                else                 VTb[(bh * ND + d) * NS + s] = bv;
            }
        }
    }
}

// ---------------------------------------------------------------------------
// Flash attention (transposed scores), one wave per (b,h,16-query tile).
// S^T = K_tile @ Q^T, online softmax along VGPR axis, ctx^T = V^T @ P^T.
// ---------------------------------------------------------------------------
__global__ __launch_bounds__(256) void attn_kernel(const __bf16* __restrict__ Qb,
                                                   const __bf16* __restrict__ Kb,
                                                   const __bf16* __restrict__ VTb,
                                                   __bf16* __restrict__ ctx) {
    const int lane = threadIdx.x & 31;
    const int wid  = threadIdx.x >> 5;
    const int id   = blockIdx.x * 8 + wid;     // 8192 = B*H*(S/16)
    const int qt = id & 127;
    const int bh = id >> 7;                    // b*16 + h
    const int q0 = qt * 16;
    const int nl = lane & 15, hl = lane >> 4;

    const __bf16* Qh = Qb  + (size_t)bh * NS * ND;
    const __bf16* Kh = Kb  + (size_t)bh * NS * ND;
    const __bf16* Vh = VTb + (size_t)bh * ND * NS;

    // Q^T B-fragments (d is the contraction dim, contiguous in memory)
    const v16bf bq0 = load_b_frag(Qh, ND, q0, 0,  lane);
    const v16bf bq1 = load_b_frag(Qh, ND, q0, 32, lane);

    const v8f z8 = {0.f,0.f,0.f,0.f,0.f,0.f,0.f,0.f};
    v8f accv[4] = { z8, z8, z8, z8 };          // ctx^T: d-chunks 0..3 (x16)
    float m_run = -3.0e38f, l_run = 0.0f;
    const float sc = 0.1803368801111388f;      // log2(e) / sqrt(64)
    const int q = q0 + nl;
    const int kend = q0 + 16;

    for (int k0 = 0; k0 < kend; k0 += 32) {
        // ---- scores: S^T tiles (sk 0-15 -> s0, sk 16-31 -> s1)
        v8f s0 = z8, s1 = z8;
        v16bf ka;
        ka = load_a_frag(Kh, ND, k0,      0,  lane); s0 = wmma_bf16(ka, bq0, s0);
        ka = load_a_frag(Kh, ND, k0,      32, lane); s0 = wmma_bf16(ka, bq1, s0);
        ka = load_a_frag(Kh, ND, k0 + 16, 0,  lane); s1 = wmma_bf16(ka, bq0, s1);
        ka = load_a_frag(Kh, ND, k0 + 16, 32, lane); s1 = wmma_bf16(ka, bq1, s1);

        // ---- scale + causal mask + tile max
        float tmax = -3.0e38f;
        #pragma unroll
        for (int r = 0; r < 8; r++) {
            const int kk0 = k0 + r + 8 * hl;
            const int kk1 = kk0 + 16;
            float v0 = s0[r] * sc;
            float v1 = s1[r] * sc;
            v0 = (kk0 > q) ? -3.0e38f : v0;
            v1 = (kk1 > q) ? -3.0e38f : v1;
            s0[r] = v0; s1[r] = v1;
            tmax = fmaxf(tmax, fmaxf(v0, v1));
        }
        tmax = fmaxf(tmax, __shfl_xor(tmax, 16, 32));
        const float m_new = fmaxf(m_run, tmax);
        const float alpha = exp2f(m_run - m_new);
        m_run = m_new;
        l_run *= alpha;
        #pragma unroll
        for (int t = 0; t < 4; t++) accv[t] = accv[t] * alpha;

        // ---- exp + partial row sums (per-lane; halves merged at the end)
        float lsum = 0.0f;
        #pragma unroll
        for (int r = 0; r < 8; r++) {
            const float e0 = exp2f(s0[r] - m_new);
            const float e1 = exp2f(s1[r] - m_new);
            s0[r] = e0; s1[r] = e1;
            lsum += e0 + e1;
        }
        l_run += lsum;

        // ---- repack P^T (f32 D-layout) into WMMA B-layout (bf16 32x16)
        v16bf pb;
        #pragma unroll
        for (int r = 0; r < 8; r++) {
            const float o0 = __shfl_xor(s0[r], 16, 32);
            const float o1 = __shfl_xor(s1[r], 16, 32);
            const float e_lo = hl ? o1 : s0[r];   // elems 0..7
            const float e_hi = hl ? s1[r] : o0;   // elems 8..15
            pb[r]     = (__bf16)e_lo;
            pb[r + 8] = (__bf16)e_hi;
        }

        // ---- ctx^T += V^T @ P^T (A-frag from V^T is per-lane contiguous)
        #pragma unroll
        for (int t = 0; t < 4; t++) {
            v16bf va = load_a_frag(Vh, NS, t * 16, k0, lane);
            accv[t] = wmma_bf16(va, pb, accv[t]);
        }
    }

    l_run += __shfl_xor(l_run, 16, 32);
    const float inv = 1.0f / l_run;

    // ---- store ctx as [B, S, E] bf16 (row = token, col = h*64 + d)
    const int hh = bh & 15, bb = bh >> 4;
    const size_t rowbase = ((size_t)bb * NS + q0 + nl) * NE + hh * ND;
    #pragma unroll
    for (int t = 0; t < 4; t++) {
        #pragma unroll
        for (int r = 0; r < 8; r++) {
            const int d = t * 16 + r + 8 * hl;
            ctx[rowbase + d] = (__bf16)(accv[t][r] * inv);
        }
    }
}

// ---------------------------------------------------------------------------
// Out-projection GEMM: attn_out[m,n] = sum_e ctx[m,e]*Wout[n,e] + bias[n] (f32)
// ---------------------------------------------------------------------------
__global__ __launch_bounds__(256) void outproj_gemm(const __bf16* __restrict__ Cx,
                                                    const __bf16* __restrict__ W,
                                                    const float* __restrict__ bias,
                                                    float* __restrict__ out) {
    const int lane = threadIdx.x & 31;
    const int wid  = threadIdx.x >> 5;
    const int tid  = blockIdx.x * 8 + wid;           // 2048 wave tiles
    const int tilesN = NE / 64;                      // 16
    const int tn = tid % tilesN, tm = tid / tilesN;
    const int m0 = tm * 64, n0 = tn * 64;

    const v8f z8 = {0.f,0.f,0.f,0.f,0.f,0.f,0.f,0.f};
    v8f acc[4][4];
    #pragma unroll
    for (int i = 0; i < 4; i++)
        #pragma unroll
        for (int j = 0; j < 4; j++) acc[i][j] = z8;

    for (int k0 = 0; k0 < NE; k0 += 32) {
        if (k0 + 32 < NE) {
            __builtin_prefetch(Cx + (size_t)(m0 + (lane & 15)) * NE + k0 + 32, 0, 1);
            __builtin_prefetch(W  + (size_t)(n0 + (lane & 15)) * NE + k0 + 32, 0, 1);
        }
        v16bf a[4], b[4];
        #pragma unroll
        for (int i = 0; i < 4; i++) a[i] = load_a_frag(Cx, NE, m0 + 16 * i, k0, lane);
        #pragma unroll
        for (int j = 0; j < 4; j++) b[j] = load_b_frag(W, NE, n0 + 16 * j, k0, lane);
        #pragma unroll
        for (int i = 0; i < 4; i++)
            #pragma unroll
            for (int j = 0; j < 4; j++)
                acc[i][j] = wmma_bf16(a[i], b[j], acc[i][j]);
    }

    const int nl = lane & 15, hl = lane >> 4;
    #pragma unroll
    for (int j = 0; j < 4; j++) {
        const int n = n0 + 16 * j + nl;
        const float bn = bias[n];
        #pragma unroll
        for (int i = 0; i < 4; i++) {
            #pragma unroll
            for (int r = 0; r < 8; r++) {
                const int m = m0 + 16 * i + r + 8 * hl;
                out[(size_t)m * NE + n] = acc[i][j][r] + bn;  // coalesced b32
            }
        }
    }
}

// ---------------------------------------------------------------------------
// Residual + LayerNorm, one wave per row of 1024 (f32 in/out).
// ---------------------------------------------------------------------------
__global__ __launch_bounds__(256) void resid_ln(const float* __restrict__ x,
                                                const float* __restrict__ attn,
                                                const float* __restrict__ gamma,
                                                const float* __restrict__ beta,
                                                float* __restrict__ out) {
    const int lane = threadIdx.x & 31;
    const int wid  = threadIdx.x >> 5;
    const int row  = blockIdx.x * 8 + wid;           // 8192 rows
    const float4* xr = (const float4*)(x    + (size_t)row * NE);
    const float4* ar = (const float4*)(attn + (size_t)row * NE);

    float4 v[8];
    float s = 0.f, sq = 0.f;
    #pragma unroll
    for (int c = 0; c < 8; c++) {
        const int idx = c * 32 + lane;               // coalesced float4
        const float4 a = xr[idx];
        const float4 b = ar[idx];
        float4 r;
        r.x = a.x + b.x; r.y = a.y + b.y; r.z = a.z + b.z; r.w = a.w + b.w;
        v[c] = r;
        s  += r.x + r.y + r.z + r.w;
        sq += r.x * r.x + r.y * r.y + r.z * r.z + r.w * r.w;
    }
    #pragma unroll
    for (int off = 16; off >= 1; off >>= 1) {
        s  += __shfl_xor(s,  off, 32);
        sq += __shfl_xor(sq, off, 32);
    }
    const float mu  = s * (1.0f / NE);
    const float var = sq * (1.0f / NE) - mu * mu;
    const float rs  = rsqrtf(var + 1e-5f);

    float4* o = (float4*)(out + (size_t)row * NE);
    const float4* g4 = (const float4*)gamma;
    const float4* b4 = (const float4*)beta;
    #pragma unroll
    for (int c = 0; c < 8; c++) {
        const int idx = c * 32 + lane;
        const float4 g = g4[idx];
        const float4 bb = b4[idx];
        const float4 r = v[c];
        float4 res;
        res.x = (r.x - mu) * rs * g.x + bb.x;
        res.y = (r.y - mu) * rs * g.y + bb.y;
        res.z = (r.z - mu) * rs * g.z + bb.z;
        res.w = (r.w - mu) * rs * g.w + bb.w;
        o[idx] = res;
    }
}

// ---------------------------------------------------------------------------
// Host launcher
// ---------------------------------------------------------------------------
extern "C" void kernel_launch(void* const* d_in, const int* in_sizes, int n_in,
                              void* d_out, int out_size, void* d_ws, size_t ws_size,
                              hipStream_t stream) {
    const float* txt    = (const float*)d_in[0];   // [4,2048,1024]
    const float* in_w   = (const float*)d_in[1];   // [3072,1024]
    const float* in_b   = (const float*)d_in[2];   // [3072]
    const float* out_w  = (const float*)d_in[3];   // [1024,1024]
    const float* out_b  = (const float*)d_in[4];   // [1024]
    const float* gamma  = (const float*)d_in[5];   // [1024]
    const float* beta   = (const float*)d_in[6];   // [1024]
    float* out = (float*)d_out;

    // Workspace layout (bytes)
    char* ws = (char*)d_ws;
    size_t off = 0;
    __bf16* Xb   = (__bf16*)(ws + off); off += (size_t)NM * NE * 2;     // 16 MB
    __bf16* Winb = (__bf16*)(ws + off); off += (size_t)N3E * NE * 2;    //  6 MB
    __bf16* Woutb= (__bf16*)(ws + off); off += (size_t)NE * NE * 2;     //  2 MB
    __bf16* Qb   = (__bf16*)(ws + off); off += (size_t)NM * NE * 2;     // 16 MB
    __bf16* Kb   = (__bf16*)(ws + off); off += (size_t)NM * NE * 2;     // 16 MB
    __bf16* VTb  = (__bf16*)(ws + off); off += (size_t)NM * NE * 2;     // 16 MB
    __bf16* Ctx  = (__bf16*)(ws + off); off += (size_t)NM * NE * 2;     // 16 MB
    float*  Aout = (float*)(ws + off);  off += (size_t)NM * NE * 4;     // 32 MB

    // 1) fp32 -> bf16 conversions
    cvt_f32_bf16_4<<<(NM * NE / 4) / 256, 256, 0, stream>>>(txt,   Xb,    NM * NE / 4);
    cvt_f32_bf16_4<<<(N3E * NE / 4) / 256, 256, 0, stream>>>(in_w, Winb,  N3E * NE / 4);
    cvt_f32_bf16_4<<<(NE * NE / 4) / 256, 256, 0, stream>>>(out_w, Woutb, NE * NE / 4);

    // 2) QKV projection (6144 wave tiles of 64x64, 8 waves / block)
    qkv_gemm<<<(NM / 64) * (N3E / 64) / 8, 256, 0, stream>>>(Xb, Winb, in_b, Qb, Kb, VTb);

    // 3) Flash attention (8192 q-tiles, 8 waves / block)
    attn_kernel<<<NB * NH * (NS / 16) / 8, 256, 0, stream>>>(Qb, Kb, VTb, Ctx);

    // 4) Output projection (2048 wave tiles)
    outproj_gemm<<<(NM / 64) * (NE / 64) / 8, 256, 0, stream>>>(Ctx, Woutb, out_b, Aout);

    // 5) Residual + LayerNorm
    resid_ln<<<NM / 8, 256, 0, stream>>>(txt, Aout, gamma, beta, out);
}
